// ExpertsLinear_87308095193289
// MI455X (gfx1250) — matrix operand order repaired and checked by reading, not compiled, not run on hardware
//
#include <hip/hip_runtime.h>

#define EXPERTS 8
#define IN_DIM  512
#define OUT_DIM 512
#define BATCH   65536
#define M_TILE  64
#define XS_STRIDE 520   // bf16 elems per LDS row; 520*2B = 1040B = 65*16 (16B aligned, bank-spread)

typedef __attribute__((ext_vector_type(16))) __bf16 v16bf;
typedef __attribute__((ext_vector_type(8)))  __bf16 bf16x8;
typedef __attribute__((ext_vector_type(8)))  float  v8f;
typedef __attribute__((ext_vector_type(4)))  float  f32x4;
typedef __attribute__((ext_vector_type(4)))  unsigned int u32x4;
typedef __attribute__((ext_vector_type(8)))  int    i32x8;
typedef __attribute__((ext_vector_type(4)))  int    i32x4;

__device__ __forceinline__ v16bf cat16(bf16x8 lo, bf16x8 hi) {
  return __builtin_shufflevector(lo, hi, 0,1,2,3,4,5,6,7,8,9,10,11,12,13,14,15);
}

// ---------- prep: W [E][K][N] f32  ->  WT_hi/WT_lo [E][N][K] bf16 (split once, L2-resident: 8 MB << 192 MB)
__global__ void w_split_transpose(const float* __restrict__ W,
                                  __bf16* __restrict__ WT_hi,
                                  __bf16* __restrict__ WT_lo) {
  int idx = blockIdx.x * blockDim.x + threadIdx.x;   // output-linear: e*N*K + n*K + k
  int k = idx & 511;
  int n = (idx >> 9) & 511;
  int e = idx >> 18;
  float f = W[((size_t)e << 18) + ((size_t)k << 9) + n];
  __bf16 h = (__bf16)f;
  __bf16 l = (__bf16)(f - (float)h);
  WT_hi[idx] = h;
  WT_lo[idx] = l;
}

// ---------- main: gated multi-expert GEMM, bf16x3 WMMA, f32 accumulate, TDM-staged x tile
__global__ __launch_bounds__(512, 1)
void moe_gemm_bf16x3(const float*  __restrict__ x,       // [B][512]
                     const float*  __restrict__ gates,   // [B][8]
                     const __bf16* __restrict__ WT_hi,   // [E][N][K]
                     const __bf16* __restrict__ WT_lo,   // [E][N][K]
                     const float*  __restrict__ bias,    // [E][1][512]
                     float*        __restrict__ out) {   // [B][512]
  extern __shared__ char smem[];
  float*  xs_f32 = (float*)smem;                               // [64][512] transient f32 tile (TDM dest)
  __bf16* xs_hi  = (__bf16*)(xs_f32 + M_TILE * IN_DIM);        // [64][XS_STRIDE]
  __bf16* xs_lo  = xs_hi + M_TILE * XS_STRIDE;                 // [64][XS_STRIDE]
  float*  gsm    = (float*)(xs_lo + M_TILE * XS_STRIDE);       // [64][8]

  const int tid  = threadIdx.x;
  const int lane = tid & 31;
  const int wave = tid >> 5;        // 0..15
  const int wm   = wave >> 3;       // 0..1 : rows wm*32 .. wm*32+31
  const int wn   = wave & 7;        // 0..7 : cols wn*64 .. wn*64+63
  const int l16  = lane & 15;
  const int half = lane >> 4;
  const int rowBase = blockIdx.x * M_TILE;

  // ---- TDM: async DMA of the 64x512 f32 x tile into LDS (one descriptor, one wave issues)
  if (wave == 0) {
    unsigned long long ga = (unsigned long long)(const void*)(x + (size_t)rowBase * IN_DIM);
    unsigned int lds_off  = (unsigned int)(unsigned long long)(void*)xs_f32;
    u32x4 g0 = { 1u,                                   // count=1 (valid user descriptor)
                 lds_off,                              // lds_addr
                 (unsigned int)(ga & 0xffffffffu),     // global_addr[31:0]
                 (unsigned int)((ga >> 32) & 0x01ffffffu) | (2u << 30) }; // addr[56:32] | type=2
    i32x8 g1 = { (int)(2u << 16),                      // data_size = 4B
                 (int)(512u << 16),                    // tensor_dim0 = 512 (bits 79:48, low half)
                 0,                                    // tensor_dim0 hi16 = 0; tensor_dim1 lo16 = 0
                 (int)((512u << 16) | 1u),             // tile_dim0 = 512 ; tensor_dim1 hi16 (65536)
                 64,                                   // tile_dim1 = 64 ; tile_dim2 = 0
                 512,                                  // tensor_dim0_stride = 512 (lo32)
                 0, 0 };                               // stride hi / dim1_stride unused (2-D)
    i32x4 gz4 = { 0, 0, 0, 0 };
    i32x8 gz8 = { 0, 0, 0, 0, 0, 0, 0, 0 };
    __builtin_amdgcn_tensor_load_to_lds(g0, g1, gz4, gz4, gz8, 0);
    __builtin_amdgcn_s_wait_tensorcnt(0);
  }
  gsm[tid] = gates[(size_t)rowBase * EXPERTS + tid];   // 64 rows x 8 experts
  __syncthreads();

  // ---- split once: LDS f32 -> (hi, lo) bf16 planes (bank-padded rows)
  #pragma unroll
  for (int i = 0; i < 16; ++i) {
    int f  = tid + i * 512;                 // float4 index over [64][128]
    int r  = f >> 7;
    int c4 = (f & 127) << 2;
    f32x4 v = *(const f32x4*)(xs_f32 + ((size_t)f << 2));
    #pragma unroll
    for (int q = 0; q < 4; ++q) {
      float  fv = v[q];
      __bf16 h  = (__bf16)fv;
      xs_hi[r * XS_STRIDE + c4 + q] = h;
      xs_lo[r * XS_STRIDE + c4 + q] = (__bf16)(fv - (float)h);
    }
  }
  __syncthreads();                          // planes ready; no barriers after this point

  v8f zero = {};
  v8f ctot[2][4], ce[2][4];
  #pragma unroll
  for (int r = 0; r < 2; ++r)
    #pragma unroll
    for (int c = 0; c < 4; ++c) { ctot[r][c] = zero; ce[r][c] = zero; }

  for (int e = 0; e < EXPERTS; ++e) {
    for (int kb = 0; kb < IN_DIM; kb += 32) {
      // A operands: per lane K = kb + 8*half + {0..7} and kb + 16 + 8*half + {0..7}
      v16bf ah[2], al[2];
      #pragma unroll
      for (int rt = 0; rt < 2; ++rt) {
        int row = wm * 32 + rt * 16 + l16;
        const __bf16* ph = xs_hi + row * XS_STRIDE + kb + 8 * half;
        const __bf16* pl = xs_lo + row * XS_STRIDE + kb + 8 * half;
        ah[rt] = cat16(*(const bf16x8*)ph, *(const bf16x8*)(ph + 16));
        al[rt] = cat16(*(const bf16x8*)pl, *(const bf16x8*)(pl + 16));
      }
      // B operands straight from L2-resident pre-split W; per lane K = kb + 8*half + {0..7} and kb + 16 + 8*half + {0..7}
      #pragma unroll
      for (int ct = 0; ct < 4; ++ct) {
        int n = wn * 64 + ct * 16 + l16;
        size_t off = (((size_t)(e * 512 + n)) << 9) + kb + 8 * half;
        v16bf bh = cat16(*(const bf16x8*)(WT_hi + off), *(const bf16x8*)(WT_hi + off + 16));
        v16bf bl = cat16(*(const bf16x8*)(WT_lo + off), *(const bf16x8*)(WT_lo + off + 16));
        #pragma unroll
        for (int rt = 0; rt < 2; ++rt) {
          ce[rt][ct] = __builtin_amdgcn_wmma_f32_16x16x32_bf16(
              false, ah[rt], false, bh, (short)0, ce[rt][ct], false, false);
          ce[rt][ct] = __builtin_amdgcn_wmma_f32_16x16x32_bf16(
              false, al[rt], false, bh, (short)0, ce[rt][ct], false, false);
          ce[rt][ct] = __builtin_amdgcn_wmma_f32_16x16x32_bf16(
              false, ah[rt], false, bl, (short)0, ce[rt][ct], false, false);
        }
      }
    }
    // fold gate (exact f32) + bias into total accumulator; C row = j + 8*half within tile
    float g[2][8];
    #pragma unroll
    for (int rt = 0; rt < 2; ++rt)
      #pragma unroll
      for (int j = 0; j < 8; ++j)
        g[rt][j] = gsm[(wm * 32 + rt * 16 + j + 8 * half) * EXPERTS + e];
    #pragma unroll
    for (int ct = 0; ct < 4; ++ct) {
      float bv = bias[(size_t)e * OUT_DIM + wn * 64 + ct * 16 + l16];
      #pragma unroll
      for (int rt = 0; rt < 2; ++rt)
        #pragma unroll
        for (int j = 0; j < 8; ++j)
          ctot[rt][ct][j] += g[rt][j] * (ce[rt][ct][j] + bv);
      #pragma unroll
      for (int rt = 0; rt < 2; ++rt) ce[rt][ct] = zero;
    }
  }

  // ---- store 64x512 f32 tile
  #pragma unroll
  for (int rt = 0; rt < 2; ++rt)
    #pragma unroll
    for (int j = 0; j < 8; ++j) {
      size_t row = (size_t)(rowBase + wm * 32 + rt * 16 + j + 8 * half);
      #pragma unroll
      for (int ct = 0; ct < 4; ++ct)
        out[row * OUT_DIM + wn * 64 + ct * 16 + l16] = ctot[rt][ct][j];
    }
}

extern "C" void kernel_launch(void* const* d_in, const int* in_sizes, int n_in,
                              void* d_out, int out_size, void* d_ws, size_t ws_size,
                              hipStream_t stream) {
  const float* x       = (const float*)d_in[0];
  const float* weights = (const float*)d_in[1];
  const float* W       = (const float*)d_in[2];
  const float* b       = (const float*)d_in[3];
  float* out = (float*)d_out;

  __bf16* WT_hi = (__bf16*)d_ws;                                  // 4 MB
  __bf16* WT_lo = WT_hi + (size_t)EXPERTS * IN_DIM * OUT_DIM;     // 4 MB

  const int wtot = EXPERTS * IN_DIM * OUT_DIM;                    // 2,097,152
  w_split_transpose<<<wtot / 256, 256, 0, stream>>>(W, WT_hi, WT_lo);

  const size_t smem = (size_t)M_TILE * IN_DIM * sizeof(float)          // 131072 f32 TDM tile
                    + (size_t)2 * M_TILE * XS_STRIDE * sizeof(__bf16)  // 133120 bf16 planes
                    + (size_t)M_TILE * EXPERTS * sizeof(float);        //   2048 gates -> 266240 B
  (void)hipFuncSetAttribute((const void*)moe_gemm_bf16x3,
                            hipFuncAttributeMaxDynamicSharedMemorySize, (int)smem);
  moe_gemm_bf16x3<<<BATCH / M_TILE, 512, smem, stream>>>(
      x, weights, WT_hi, WT_lo, b, out);
}
